// RelPartialLearnableMultiHeadAttn_35021163332269
// MI455X (gfx1250) — matrix operand order repaired and checked by reading, not compiled, not run on hardware
//
#include <hip/hip_runtime.h>
#include <hip/hip_bf16.h>

// ---------------------------------------------------------------------------
// Transformer-XL relative multi-head attention, fused, bf16 WMMA on gfx1250.
// ---------------------------------------------------------------------------

#define QLEN   1024
#define BSZ    8
#define NHEAD  16
#define DHEAD  64
#define DMODEL 1024
#define SCALE  0.125f   // 1/sqrt(64)
#define NEGINF (-3.0e38f)

typedef __attribute__((ext_vector_type(16))) __bf16 bf16x16;
typedef __attribute__((ext_vector_type(8)))  float  v8f;

// --------------------- async global -> LDS copy (CDNA5) --------------------
#if __has_builtin(__builtin_amdgcn_global_load_async_to_lds_b128) && \
    __has_builtin(__builtin_amdgcn_s_wait_asynccnt)
#define HAVE_ASYNC_LDS 1
#else
#define HAVE_ASYNC_LDS 0
#endif

typedef __attribute__((__vector_size__(16))) int i32x4_t;

__device__ __forceinline__ void copy16_g2l(void* lds, const void* gm) {
#if HAVE_ASYNC_LDS
    // GLOBAL_LOAD_ASYNC_TO_LDS_B128: ASYNCcnt-tracked, no VGPR round trip
    __builtin_amdgcn_global_load_async_to_lds_b128(
        (__attribute__((address_space(1))) i32x4_t*)(uintptr_t)gm,
        (__attribute__((address_space(3))) i32x4_t*)(uint32_t)(uintptr_t)lds,
        0, 0);
#else
    *(uint4*)lds = *(const uint4*)gm;
#endif
}

__device__ __forceinline__ void async_fence() {
#if HAVE_ASYNC_LDS
    __builtin_amdgcn_s_wait_asynccnt(0);   // s_wait_asynccnt 0
#endif
}

__device__ __forceinline__ v8f wmma_bf16(bf16x16 a, bf16x16 b, v8f c) {
    // D = A(16x32) * B(32x16) + C, f32 accumulate
    return __builtin_amdgcn_wmma_f32_16x16x32_bf16(
        /*neg_a=*/false, a, /*neg_b=*/false, b,
        /*c_mod=*/(short)0, c, /*reuse_a=*/false, /*reuse_b=*/false);
}

// A-matrix fragment (16x32 bf16), per ISA 7.12.2:
// lane m = lane&15 selects row; half hh: K = kbase + hh + 8*hi (+8 if hh>=8)
__device__ __forceinline__ bf16x16 load_a_frag(const __bf16* row, int kbase, int hi) {
    bf16x16 a;
    const __bf16* p0 = row + kbase + 8 * hi;
    const __bf16* p1 = row + kbase + 16 + 8 * hi;
#pragma unroll
    for (int t = 0; t < 8; ++t) { a[t] = p0[t]; a[t + 8] = p1[t]; }
    return a;
}

// B-matrix fragment (32x16 bf16) loaded from N-major (N x K) storage:
// lane n = lane&15 selects column; K = kbase + hh + 8*hi (+8 if hh>=8)
__device__ __forceinline__ bf16x16 load_b_frag(const __bf16* nrow, int kbase, int hi) {
    bf16x16 b;
    const __bf16* p0 = nrow + kbase + 8 * hi;
    const __bf16* p1 = nrow + kbase + 16 + 8 * hi;
#pragma unroll
    for (int t = 0; t < 8; ++t) { b[t] = p0[t]; b[t + 8] = p1[t]; }
    return b;
}

__device__ __forceinline__ v8f vzero8() {
    v8f z = {0.f, 0.f, 0.f, 0.f, 0.f, 0.f, 0.f, 0.f};
    return z;
}

// ---------------------------------------------------------------------------
// Generic tiled GEMM: C[M,N] = A[M,K] (f32 or bf16) * B[K,N] (f32)
// Block 256 thr = 8 waves; tile 128x128, K-step 32; wave grid 4x2, 32x64/wave,
// 8 WMMAs per wave per k-step. Software prefetch of next k-step tiles.
// MODE 0: scatter QKV -> Q/K/V [b][h][q][d] bf16
// MODE 1: scatter Rk  -> [h][m][d] bf16
// MODE 2: plain f32 row-major [M][N]
// ---------------------------------------------------------------------------
template <typename TA, int MODE>
__global__ __launch_bounds__(256)
void gemm_kernel(const TA* __restrict__ A, const float* __restrict__ B,
                 void* __restrict__ C0, void* __restrict__ C1, void* __restrict__ C2,
                 int M, int N, int K) {
    __shared__ __align__(16) __bf16 As[128][32];
    __shared__ __align__(16) __bf16 Bts[128][32];  // N-major (n, k)

    const int tid  = threadIdx.x;
    const int wid  = tid >> 5;
    const int lane = tid & 31;
    const int ln   = lane & 15;
    const int hi   = lane >> 4;
    const int wm   = wid >> 1;      // 0..3  (M direction, 32 rows each)
    const int wn   = wid & 1;       // 0..1  (N direction, 64 cols each)
    const int m0   = blockIdx.x * 128;
    const int n0   = blockIdx.y * 128;

    v8f acc[2][4];
#pragma unroll
    for (int ti = 0; ti < 2; ++ti)
#pragma unroll
        for (int tj = 0; tj < 4; ++tj) acc[ti][tj] = vzero8();

    for (int kb = 0; kb < K; kb += 32) {
        // stage A tile (128x32) as bf16: 16 contiguous elems / thread
#pragma unroll
        for (int i = 0; i < 16; ++i) {
            int idx = tid * 16 + i;           // 4096 elements
            int rr = idx >> 5, cc = idx & 31;
            As[rr][cc] = (__bf16)((float)A[(size_t)(m0 + rr) * K + kb + cc]);
        }
        // stage B tile (32x128) transposed to (n,k): 16 contiguous n / thread
#pragma unroll
        for (int i = 0; i < 16; ++i) {
            int idx = tid * 16 + i;           // 4096 elements
            int kk = idx >> 7, nn = idx & 127;
            Bts[nn][kk] = (__bf16)B[(size_t)(kb + kk) * N + n0 + nn];
        }
        // prefetch next k-step tiles into cache (global_prefetch_b8)
        if (kb + 32 < K) {
            __builtin_prefetch(&A[(size_t)(m0 + (tid >> 1)) * K + kb + 32 + (tid & 1) * 16], 0, 2);
            __builtin_prefetch(&B[(size_t)(kb + 32 + (tid >> 3)) * N + n0 + (tid & 7) * 16], 0, 2);
        }
        __syncthreads();

        bf16x16 af[2], bf[4];
#pragma unroll
        for (int ti = 0; ti < 2; ++ti)
            af[ti] = load_a_frag(&As[wm * 32 + ti * 16 + ln][0], 0, hi);
#pragma unroll
        for (int tj = 0; tj < 4; ++tj)
            bf[tj] = load_b_frag(&Bts[wn * 64 + tj * 16 + ln][0], 0, hi);
#pragma unroll
        for (int ti = 0; ti < 2; ++ti)
#pragma unroll
            for (int tj = 0; tj < 4; ++tj)
                acc[ti][tj] = wmma_bf16(af[ti], bf[tj], acc[ti][tj]);
        __syncthreads();
    }

    // epilogue: C layout per ISA — lane holds row e+8*hi, col lane&15
#pragma unroll
    for (int ti = 0; ti < 2; ++ti) {
#pragma unroll
        for (int tj = 0; tj < 4; ++tj) {
#pragma unroll
            for (int e = 0; e < 8; ++e) {
                int gr = m0 + wm * 32 + ti * 16 + e + 8 * hi;
                int gc = n0 + wn * 64 + tj * 16 + ln;
                float v = acc[ti][tj][e];
                if (MODE == 0) {
                    int q = gr >> 3, b = gr & 7;
                    int part = gc >> 10, rem = gc & 1023;
                    int h = rem >> 6, d = rem & 63;
                    __bf16* dst = (__bf16*)(part == 0 ? C0 : (part == 1 ? C1 : C2));
                    dst[(((size_t)b * NHEAD + h) * QLEN + q) * DHEAD + d] = (__bf16)v;
                } else if (MODE == 1) {
                    int h = gc >> 6, d = gc & 63;
                    ((__bf16*)C0)[((size_t)h * QLEN + gr) * DHEAD + d] = (__bf16)v;
                } else {
                    ((float*)C0)[(size_t)gr * N + gc] = v;
                }
            }
        }
    }
}

// ---------------------------------------------------------------------------
// Fused relative attention. Grid (qtile=16, head=16, batch=8), 128 thr/4 waves.
// Each wave owns 16 query rows; flash-style online softmax over key tiles of 64.
// BD term: block-shared 128-row Rk band staged in LDS; per-wave unshifted GEMM
// over its 79-row sub-band, shifted via LDS gather.
// K / Q / Rk-band staging uses GLOBAL_LOAD_ASYNC_TO_LDS_B128 when available.
// ---------------------------------------------------------------------------
__global__ __launch_bounds__(128)
void attn_kernel(const __bf16* __restrict__ Q, const __bf16* __restrict__ K,
                 const __bf16* __restrict__ V, const __bf16* __restrict__ Rk,
                 const float* __restrict__ rwb, const float* __restrict__ rrb,
                 __bf16* __restrict__ AV) {
    // QPs: Q tile during setup, then per-wave P staging (same footprint)
    __shared__ __align__(16) __bf16 QPs[64][64];
    __shared__ __align__(16) __bf16 Ks[64][64];
    __shared__ __align__(16) __bf16 Vts[64][64];      // transposed: [d][key]
    __shared__ __align__(16) __bf16 Rband[128][64];   // block-wide Rk band
    __shared__ __align__(16) float  Gs[4][16][80];    // per-wave BD band

    const int qt = blockIdx.x, h = blockIdx.y, b = blockIdx.z;
    const int tid = threadIdx.x, wid = tid >> 5, lane = tid & 31;
    const int ln = lane & 15, hi = lane >> 4;
    const int Iblk = qt * 64;
    const int Iw = Iblk + wid * 16;
    const int wb = 48 - 16 * wid;   // wave's offset into the block Rk band

    const size_t bh = ((size_t)b * NHEAD + h) * QLEN * DHEAD;
    const __bf16* Qbh = Q + bh;
    const __bf16* Kbh = K + bh;
    const __bf16* Vbh = V + bh;
    const __bf16* Rkh = Rk + (size_t)h * QLEN * DHEAD;

    // stage Q tile (64x64 halves = 512 x b128), async when available
    {
        const uint4* src = (const uint4*)(Qbh + (size_t)Iblk * DHEAD);
        uint4* dst = (uint4*)&QPs[0][0];
#pragma unroll
        for (int i = 0; i < 4; ++i)
            copy16_g2l(dst + tid + i * 128, src + tid + i * 128);
    }
    async_fence();
    __syncthreads();

    // per-wave A fragments with r_w / r_r biases folded in (f32 add, bf16 store)
    bf16x16 aw[2], ar[2];
    {
        const __bf16* qrow = &QPs[wid * 16 + ln][0];
        const float* rwbh = rwb + h * DHEAD;
        const float* rrbh = rrb + h * DHEAD;
#pragma unroll
        for (int kk = 0; kk < 2; ++kk) {
#pragma unroll
            for (int hh = 0; hh < 16; ++hh) {
                int d = kk * 32 + hh + 8 * hi + (hh >= 8 ? 8 : 0);
                float f = (float)qrow[d];
                aw[kk][hh] = (__bf16)(f + rwbh[d]);
                ar[kk][hh] = (__bf16)(f + rrbh[d]);
            }
        }
    }

    float rowm[8], rowl[8];
    v8f accO[4];
#pragma unroll
    for (int e = 0; e < 8; ++e) { rowm[e] = NEGINF; rowl[e] = 0.f; }
#pragma unroll
    for (int dd = 0; dd < 4; ++dd) accO[dd] = vzero8();

    for (int J = 0; J <= Iblk; J += 64) {
        const int row0 = J + (QLEN - 1) - Iblk - 63;  // block band start
        // stage K tile (async), V tile (manual transpose), Rk band (async)
        {
            const uint4* src = (const uint4*)(Kbh + (size_t)J * DHEAD);
            uint4* dst = (uint4*)&Ks[0][0];
#pragma unroll
            for (int i = 0; i < 4; ++i)
                copy16_g2l(dst + tid + i * 128, src + tid + i * 128);
            const __bf16* vsrc = Vbh + (size_t)J * DHEAD;
#pragma unroll
            for (int i = 0; i < 4; ++i) {
                int idx = tid + i * 128;          // uint4 index, 512 total
                int key = idx >> 3, d0 = (idx & 7) * 8;
                uint4 pk = ((const uint4*)vsrc)[idx];
                const __bf16* hp = (const __bf16*)&pk;
#pragma unroll
                for (int t = 0; t < 8; ++t) Vts[d0 + t][key] = hp[t];
            }
            // Rk band: 128 rows x 64 halves = 1024 x b128, 8 per thread
#pragma unroll
            for (int i = 0; i < 8; ++i) {
                int idx = tid + i * 128;
                int rr = idx >> 3, seg = idx & 7;
                int srow = row0 + rr;
                srow = srow < 0 ? 0 : (srow > QLEN - 1 ? QLEN - 1 : srow);
                copy16_g2l((uint4*)&Rband[rr][0] + seg,
                           (const uint4*)(Rkh + (size_t)srow * DHEAD) + seg);
            }
        }
        // prefetch next tile's K/V/Rk lines while this tile is consumed
        if (J + 64 <= Iblk) {
            if (tid < 64) {
                __builtin_prefetch((const char*)(Kbh + (size_t)(J + 64) * DHEAD) + tid * 128, 0, 2);
                __builtin_prefetch((const char*)(Vbh + (size_t)(J + 64) * DHEAD) + tid * 128, 0, 2);
            }
            int prow = row0 + 64 + tid;
            prow = prow < 0 ? 0 : (prow > QLEN - 1 ? QLEN - 1 : prow);
            __builtin_prefetch(Rkh + (size_t)prow * DHEAD, 0, 2);
        }
        async_fence();
        __syncthreads();

        // BD band: G[di,t] = (q+rr) . Rk[mbase+t], t in [0,80)  (from LDS band)
#pragma unroll
        for (int tt = 0; tt < 5; ++tt) {
            const __bf16* bt = &Rband[wb + tt * 16 + ln][0];
            bf16x16 b0 = load_b_frag(bt, 0, hi);
            bf16x16 b1 = load_b_frag(bt, 32, hi);
            v8f g = wmma_bf16(ar[0], b0, vzero8());
            g = wmma_bf16(ar[1], b1, g);
#pragma unroll
            for (int e = 0; e < 8; ++e)
                Gs[wid][e + 8 * hi][tt * 16 + ln] = g[e];
        }
        __syncthreads();   // make G visible across lanes of the wave

        // AC scores + shifted BD + causal mask
        float sv[4][8];
#pragma unroll
        for (int jj = 0; jj < 4; ++jj) {
            const __bf16* bt = &Ks[jj * 16 + ln][0];
            bf16x16 b0 = load_b_frag(bt, 0, hi);
            bf16x16 b1 = load_b_frag(bt, 32, hi);
            v8f s = wmma_bf16(aw[0], b0, vzero8());
            s = wmma_bf16(aw[1], b1, s);
#pragma unroll
            for (int e = 0; e < 8; ++e) {
                int r = e + 8 * hi;
                int i = Iw + r;
                int j = J + jj * 16 + ln;
                float bd = Gs[wid][r][jj * 16 + ln + 15 - r];
                float val = SCALE * (s[e] + bd);
                sv[jj][e] = (j <= i) ? val : NEGINF;
            }
        }

        // online softmax (row reductions across the 16-lane half-wave)
#pragma unroll
        for (int e = 0; e < 8; ++e) {
            float mx = fmaxf(fmaxf(sv[0][e], sv[1][e]), fmaxf(sv[2][e], sv[3][e]));
#pragma unroll
            for (int off = 1; off < 16; off <<= 1)
                mx = fmaxf(mx, __shfl_xor(mx, off, 32));
            float mnew = fmaxf(rowm[e], mx);
            float corr = __expf(rowm[e] - mnew);
            rowm[e] = mnew;
            float sum = 0.f;
#pragma unroll
            for (int jj = 0; jj < 4; ++jj) {
                float p = __expf(sv[jj][e] - mnew);
                QPs[wid * 16 + e + 8 * hi][jj * 16 + ln] = (__bf16)p;  // P staging
                sum += p;
            }
#pragma unroll
            for (int off = 1; off < 16; off <<= 1)
                sum += __shfl_xor(sum, off, 32);
            rowl[e] = rowl[e] * corr + sum;
#pragma unroll
            for (int dd = 0; dd < 4; ++dd) accO[dd][e] *= corr;
        }
        __syncthreads();   // make P visible across lanes of the wave

        // O += P @ V (P reloaded in A-fragment layout from LDS)
        bf16x16 pa0 = load_a_frag(&QPs[wid * 16 + ln][0], 0, hi);
        bf16x16 pa1 = load_a_frag(&QPs[wid * 16 + ln][0], 32, hi);
#pragma unroll
        for (int dd = 0; dd < 4; ++dd) {
            const __bf16* bt = &Vts[dd * 16 + ln][0];
            bf16x16 b0 = load_b_frag(bt, 0, hi);
            bf16x16 b1 = load_b_frag(bt, 32, hi);
            accO[dd] = wmma_bf16(pa0, b0, accO[dd]);
            accO[dd] = wmma_bf16(pa1, b1, accO[dd]);
        }
        __syncthreads();   // before next tile overwrites Ks/Vts/Rband
    }

    // normalize and store attn_vec [q][b][h*64+d] as bf16
#pragma unroll
    for (int dd = 0; dd < 4; ++dd) {
#pragma unroll
        for (int e = 0; e < 8; ++e) {
            int r = e + 8 * hi;
            int q = Iw + r;
            float v = accO[dd][e] / rowl[e];
            AV[((size_t)q * BSZ + b) * DMODEL + h * DHEAD + dd * 16 + ln] = (__bf16)v;
        }
    }
}

// ---------------------------------------------------------------------------
// Residual + LayerNorm: out = LN(w + attn_out) * g + b, one block per row.
// ---------------------------------------------------------------------------
__global__ __launch_bounds__(256)
void ln_kernel(const float* __restrict__ w, const float* __restrict__ ao,
               const float* __restrict__ g, const float* __restrict__ beta,
               float* __restrict__ out) {
    const int row = blockIdx.x;
    const int tid = threadIdx.x, lane = tid & 31, wid = tid >> 5;
    const float* xw = w + (size_t)row * DMODEL;
    const float* xa = ao + (size_t)row * DMODEL;

    float x[4];
    float s = 0.f, s2 = 0.f;
#pragma unroll
    for (int i = 0; i < 4; ++i) {
        int c = tid + i * 256;
        x[i] = xw[c] + xa[c];
        s += x[i];
        s2 += x[i] * x[i];
    }
#pragma unroll
    for (int off = 16; off; off >>= 1) {
        s  += __shfl_xor(s, off, 32);
        s2 += __shfl_xor(s2, off, 32);
    }
    __shared__ float rs[8], rs2[8];
    if (lane == 0) { rs[wid] = s; rs2[wid] = s2; }
    __syncthreads();
    float S = 0.f, S2 = 0.f;
#pragma unroll
    for (int i = 0; i < 8; ++i) { S += rs[i]; S2 += rs2[i]; }
    float mu = S * (1.0f / DMODEL);
    float var = S2 * (1.0f / DMODEL) - mu * mu;
    float inv = rsqrtf(var + 1e-5f);
#pragma unroll
    for (int i = 0; i < 4; ++i) {
        int c = tid + i * 256;
        out[(size_t)row * DMODEL + c] = (x[i] - mu) * inv * g[c] + beta[c];
    }
}

// ---------------------------------------------------------------------------
extern "C" void kernel_launch(void* const* d_in, const int* in_sizes, int n_in,
                              void* d_out, int out_size, void* d_ws, size_t ws_size,
                              hipStream_t stream) {
    const float* w    = (const float*)d_in[0];   // [1024,8,1024]
    const float* r    = (const float*)d_in[1];   // [1024,1024]
    const float* rwb  = (const float*)d_in[2];   // [16,64]
    const float* rrb  = (const float*)d_in[3];   // [16,64]
    const float* Wqkv = (const float*)d_in[4];   // [1024,3072]
    const float* Wr   = (const float*)d_in[5];   // [1024,1024]
    const float* Wo   = (const float*)d_in[6];   // [1024,1024]
    const float* lng  = (const float*)d_in[7];
    const float* lnb  = (const float*)d_in[8];
    // d_in[9] = attn_mask (causal) — computed analytically, unused
    float* out = (float*)d_out;

    const size_t PH = (size_t)BSZ * NHEAD * QLEN * DHEAD;   // 8.4M halves
    __bf16* Qd  = (__bf16*)d_ws;
    __bf16* Kd  = Qd + PH;
    __bf16* Vd  = Kd + PH;
    __bf16* Rkd = Vd + PH;                                   // 16*1024*64
    __bf16* AVd = Rkd + (size_t)NHEAD * QLEN * DHEAD;        // 8192*1024
    float*  AOd = (float*)(AVd + (size_t)QLEN * BSZ * DMODEL);

    // 1) QKV projection: [8192,1024] x [1024,3072] -> scatter Q/K/V bf16
    gemm_kernel<float, 0><<<dim3(8192 / 128, 3072 / 128), 256, 0, stream>>>(
        w, Wqkv, Qd, Kd, Vd, 8192, 3072, 1024);
    // 2) Positional projection: [1024,1024] x [1024,1024] -> Rk[h][m][d] bf16
    gemm_kernel<float, 1><<<dim3(1024 / 128, 1024 / 128), 256, 0, stream>>>(
        r, Wr, Rkd, nullptr, nullptr, 1024, 1024, 1024);
    // 3) Fused relative attention -> attn_vec bf16 [q][b][n*d]
    attn_kernel<<<dim3(16, 16, 8), 128, 0, stream>>>(Qd, Kd, Vd, Rkd, rwb, rrb, AVd);
    // 4) Output projection: [8192,1024](bf16) x [1024,1024] -> f32
    gemm_kernel<__bf16, 2><<<dim3(8192 / 128, 1024 / 128), 256, 0, stream>>>(
        AVd, Wo, AOd, nullptr, nullptr, 8192, 1024, 1024);
    // 5) Residual + LayerNorm -> d_out f32
    ln_kernel<<<8192, 256, 0, stream>>>(w, AOd, lng, lnb, out);
}
